// FaReG_43860206027256
// MI455X (gfx1250) — matrix operand order repaired and checked by reading, not compiled, not run on hardware
//
#include <hip/hip_runtime.h>

#define N_ROWS 65536
#define F_DIM  4096
#define H_DIM  64
#define L_DIM  32
#define DELTA  0.7f
#define SLOPE  0.2f

typedef __attribute__((ext_vector_type(16))) __bf16 v16bf;
typedef __attribute__((ext_vector_type(8)))  float  v8f;
typedef __attribute__((ext_vector_type(4)))  float  f32x4;

__device__ __forceinline__ unsigned short f2bf(float f) {
    unsigned int u = __float_as_uint(f);
    u += 0x7FFFu + ((u >> 16) & 1u);      // round-to-nearest-even
    return (unsigned short)(u >> 16);
}
__device__ __forceinline__ float lrelu(float x) { return x > 0.f ? x : SLOPE * x; }

// pack two f32 into one dword of two bf16 (truncation) : dst = {hi[31:16], lo[31:16]}
__device__ __forceinline__ unsigned int bfpack(float lo, float hi) {
#if defined(__has_builtin) && __has_builtin(__builtin_amdgcn_perm)
    return __builtin_amdgcn_perm(__float_as_uint(hi), __float_as_uint(lo), 0x07060302u);
#else
    return (__float_as_uint(hi) & 0xFFFF0000u) | (__float_as_uint(lo) >> 16);
#endif
}

union AFrag {
    v16bf v;
    unsigned int d[8];
};

__device__ __forceinline__ void cvt_frag(AFrag& a, f32x4 f0, f32x4 f1, f32x4 f2, f32x4 f3) {
    a.d[0] = bfpack(f0.x, f0.y);
    a.d[1] = bfpack(f0.z, f0.w);
    a.d[2] = bfpack(f1.x, f1.y);
    a.d[3] = bfpack(f1.z, f1.w);
    a.d[4] = bfpack(f2.x, f2.y);
    a.d[5] = bfpack(f2.z, f2.w);
    a.d[6] = bfpack(f3.x, f3.y);
    a.d[7] = bfpack(f3.z, f3.w);
}

// ---------------------------------------------------------------------------
// Kernel 1: repack enc_w1 (F x H, f32 row-major) into bf16 in WMMA B-fragment
// order: pk[(((c*4+t)*32 + lane)*16 + e] = bf16(w1[k*H + n])
//   k = c*32 + (lane>>4)*8 + (e&7) + (e>>3)*16 ,  n = (lane&15) + 16*t
// ---------------------------------------------------------------------------
__global__ void pack_w1_kernel(const float* __restrict__ w1,
                               unsigned short* __restrict__ pk,
                               float* __restrict__ scal) {
    int idx = blockIdx.x * 256 + threadIdx.x;      // 0 .. F*H-1
    int e = idx & 15;
    int l = (idx >> 4) & 31;
    int t = (idx >> 9) & 3;
    int c = idx >> 11;
    int k = c * 32 + ((l >> 4) << 3) + (e & 7) + ((e >> 3) << 4);
    int n = (l & 15) + (t << 4);
    pk[idx] = f2bf(w1[k * H_DIM + n]);
    if (idx < 8) scal[idx] = 0.f;
}

// ---------------------------------------------------------------------------
// Kernel 2: fused VAE forward. 256 blocks x 256 threads (8 waves).
// Each wave: 32 rows = two 16-row stripes sharing 4 B fragments,
// 8 x wmma_16x16x32_bf16 per K-step, K-loop of 128 steps.
// Then 256 threads finish layers 2..7, one row each, from LDS.
// ---------------------------------------------------------------------------
__global__ void __launch_bounds__(256) vae_main_kernel(
    const float* __restrict__ x,   const float* __restrict__ eps,
    const unsigned short* __restrict__ w1p, const float* __restrict__ b1,
    const float* __restrict__ w2,  const float* __restrict__ b2,
    const float* __restrict__ muw, const float* __restrict__ mub,
    const float* __restrict__ lvw, const float* __restrict__ lvb,
    const float* __restrict__ dw1, const float* __restrict__ db1,
    const float* __restrict__ dw2, const float* __restrict__ db2,
    const float* __restrict__ dw3, const float* __restrict__ db3,
    float* __restrict__ out, float* __restrict__ prob,
    float* __restrict__ blocksum)
{
    __shared__ float sH[256 * 64];                 // h1 tile (64 KB)
    __shared__ float sW2[64 * 32], sD2[32 * 64];
    __shared__ float sMW[32 * 4], sLW[32 * 4], sD1[4 * 32];
    __shared__ float sB1[64], sD3[64], sDB2[64];
    __shared__ float sB2[32], sDB1[32];
    __shared__ float sMB[4], sLB[4], sDB3[1];
    __shared__ float sRed[256];

    const int tid = threadIdx.x;

    // stage epilogue weights
    for (int i = tid; i < 64 * 32; i += 256) sW2[i] = w2[i];
    for (int i = tid; i < 32 * 64; i += 256) sD2[i] = dw2[i];
    if (tid < 128) { sMW[tid] = muw[tid]; sLW[tid] = lvw[tid]; sD1[tid] = dw1[tid]; }
    if (tid < 64)  { sB1[tid] = b1[tid]; sD3[tid] = dw3[tid]; sDB2[tid] = db2[tid]; }
    if (tid < 32)  { sB2[tid] = b2[tid]; sDB1[tid] = db1[tid]; }
    if (tid < 4)   { sMB[tid] = mub[tid]; sLB[tid] = lvb[tid]; }
    if (tid == 0)  sDB3[0] = db3[0];

    const int lane = tid & 31;
    const int wv   = tid >> 5;
    const int rowBase = blockIdx.x * 256 + wv * 32;

    // A-fragment addressing (16-bit A 16x32 layout)
    const int mA    = lane & 15;
    const int koffA = (lane >> 4) << 3;            // 0 or 8
    const float* xr0 = x + (size_t)(rowBase + mA) * F_DIM;
    const float* xr1 = xr0 + (size_t)16 * F_DIM;

    v8f acc0 = {}, acc1 = {}, acc2 = {}, acc3 = {};   // stripe 0 (rows 0..15)
    v8f acc4 = {}, acc5 = {}, acc6 = {}, acc7 = {};   // stripe 1 (rows 16..31)
    const v16bf* bbase = ((const v16bf*)w1p) + lane;

    for (int kb = 0; kb < F_DIM / 32; ++kb) {
        const int kbase = kb * 32;
        // ---- B fragments: pre-packed, contiguous per lane, shared by stripes
        const v16bf* bp = bbase + (size_t)kb * 128;
        v16bf b0 = bp[0], b1f = bp[32], b2f = bp[64], b3f = bp[96];
        // ---- A fragments: non-temporal stream loads, perm-packed to bf16 ----
        const f32x4* g00 = (const f32x4*)(xr0 + kbase + koffA);
        const f32x4* g01 = (const f32x4*)(xr0 + kbase + 16 + koffA);
        const f32x4* g10 = (const f32x4*)(xr1 + kbase + koffA);
        const f32x4* g11 = (const f32x4*)(xr1 + kbase + 16 + koffA);
        f32x4 u0 = __builtin_nontemporal_load(g00);
        f32x4 u1 = __builtin_nontemporal_load(g00 + 1);
        f32x4 u2 = __builtin_nontemporal_load(g01);
        f32x4 u3 = __builtin_nontemporal_load(g01 + 1);
        f32x4 w0 = __builtin_nontemporal_load(g10);
        f32x4 w1_ = __builtin_nontemporal_load(g10 + 1);
        f32x4 w2_ = __builtin_nontemporal_load(g11);
        f32x4 w3 = __builtin_nontemporal_load(g11 + 1);
        AFrag a0, a1;
        cvt_frag(a0, u0, u1, u2, u3);
        cvt_frag(a1, w0, w1_, w2_, w3);

        acc0 = __builtin_amdgcn_wmma_f32_16x16x32_bf16(false, a0.v, false, b0,  (short)0, acc0, false, false);
        acc1 = __builtin_amdgcn_wmma_f32_16x16x32_bf16(false, a0.v, false, b1f, (short)0, acc1, false, false);
        acc2 = __builtin_amdgcn_wmma_f32_16x16x32_bf16(false, a0.v, false, b2f, (short)0, acc2, false, false);
        acc3 = __builtin_amdgcn_wmma_f32_16x16x32_bf16(false, a0.v, false, b3f, (short)0, acc3, false, false);
        acc4 = __builtin_amdgcn_wmma_f32_16x16x32_bf16(false, a1.v, false, b0,  (short)0, acc4, false, false);
        acc5 = __builtin_amdgcn_wmma_f32_16x16x32_bf16(false, a1.v, false, b1f, (short)0, acc5, false, false);
        acc6 = __builtin_amdgcn_wmma_f32_16x16x32_bf16(false, a1.v, false, b2f, (short)0, acc6, false, false);
        acc7 = __builtin_amdgcn_wmma_f32_16x16x32_bf16(false, a1.v, false, b3f, (short)0, acc7, false, false);
    }

    // ---- write h1 tiles (+bias, leaky) to LDS per C-layout ----
    {
        const int nC    = lane & 15;
        const int mBase = (lane >> 4) << 3;        // 0 or 8
        #pragma unroll
        for (int i = 0; i < 8; ++i) {
            const int r0 = (wv * 32 + mBase + i) * 64;
            const int r1 = r0 + 16 * 64;
            sH[r0 + nC +  0] = lrelu(acc0[i] + sB1[nC +  0]);
            sH[r0 + nC + 16] = lrelu(acc1[i] + sB1[nC + 16]);
            sH[r0 + nC + 32] = lrelu(acc2[i] + sB1[nC + 32]);
            sH[r0 + nC + 48] = lrelu(acc3[i] + sB1[nC + 48]);
            sH[r1 + nC +  0] = lrelu(acc4[i] + sB1[nC +  0]);
            sH[r1 + nC + 16] = lrelu(acc5[i] + sB1[nC + 16]);
            sH[r1 + nC + 32] = lrelu(acc6[i] + sB1[nC + 32]);
            sH[r1 + nC + 48] = lrelu(acc7[i] + sB1[nC + 48]);
        }
    }
    __syncthreads();

    // ---- per-row epilogue: layers 2..7 (one row per thread) ----
    const int row = blockIdx.x * 256 + tid;
    const float4* h4 = (const float4*)&sH[tid * 64];

    // h2 = leaky(h1 @ w2 + b2)            (64 x 32)
    float h2[32];
    {
        const float4* b4 = (const float4*)sB2;
        #pragma unroll
        for (int q = 0; q < 8; ++q) {
            const float4 b = b4[q];
            h2[4*q+0] = b.x; h2[4*q+1] = b.y; h2[4*q+2] = b.z; h2[4*q+3] = b.w;
        }
    }
    for (int i4 = 0; i4 < 16; ++i4) {
        const float4 hq = h4[i4];
        const float hv[4] = {hq.x, hq.y, hq.z, hq.w};
        #pragma unroll
        for (int s = 0; s < 4; ++s) {
            const float4* wr = (const float4*)(sW2 + (i4 * 4 + s) * 32);
            #pragma unroll
            for (int q = 0; q < 8; ++q) {
                const float4 wq = wr[q];
                h2[4*q+0] += hv[s] * wq.x;
                h2[4*q+1] += hv[s] * wq.y;
                h2[4*q+2] += hv[s] * wq.z;
                h2[4*q+3] += hv[s] * wq.w;
            }
        }
    }
    #pragma unroll
    for (int j = 0; j < 32; ++j) h2[j] = lrelu(h2[j]);

    // mu / logvar (32 x 4 each)
    float mu[4], lv[4];
    #pragma unroll
    for (int c = 0; c < 4; ++c) { mu[c] = sMB[c]; lv[c] = sLB[c]; }
    for (int j = 0; j < 32; ++j) {
        const float hv = h2[j];
        const float4 mw = ((const float4*)sMW)[j];
        const float4 lw = ((const float4*)sLW)[j];
        mu[0] += hv * mw.x; mu[1] += hv * mw.y; mu[2] += hv * mw.z; mu[3] += hv * mw.w;
        lv[0] += hv * lw.x; lv[1] += hv * lw.y; lv[2] += hv * lw.z; lv[3] += hv * lw.w;
    }
    float z[4];
    #pragma unroll
    for (int c = 0; c < 4; ++c) {
        z[c] = mu[c] + lv[c] * eps[row * 4 + c];
        out[N_ROWS     + row * 4 + c] = mu[c];
        out[5 * N_ROWS + row * 4 + c] = lv[c];
    }

    // d1 = leaky(z @ dec_w1 + db1)        (4 x 32)
    float d1[32];
    {
        const float4* b4 = (const float4*)sDB1;
        #pragma unroll
        for (int q = 0; q < 8; ++q) {
            const float4 b = b4[q];
            d1[4*q+0] = b.x; d1[4*q+1] = b.y; d1[4*q+2] = b.z; d1[4*q+3] = b.w;
        }
    }
    #pragma unroll
    for (int c = 0; c < 4; ++c) {
        const float zc = z[c];
        const float4* wr = (const float4*)(sD1 + c * 32);
        #pragma unroll
        for (int q = 0; q < 8; ++q) {
            const float4 wq = wr[q];
            d1[4*q+0] += zc * wq.x;
            d1[4*q+1] += zc * wq.y;
            d1[4*q+2] += zc * wq.z;
            d1[4*q+3] += zc * wq.w;
        }
    }
    #pragma unroll
    for (int j = 0; j < 32; ++j) d1[j] = lrelu(d1[j]);

    // d2 = d1 @ dec_w2 + db2              (32 x 64)
    float d2[64];
    {
        const float4* b4 = (const float4*)sDB2;
        #pragma unroll
        for (int q = 0; q < 16; ++q) {
            const float4 b = b4[q];
            d2[4*q+0] = b.x; d2[4*q+1] = b.y; d2[4*q+2] = b.z; d2[4*q+3] = b.w;
        }
    }
    for (int j = 0; j < 32; ++j) {
        const float dv = d1[j];
        const float4* wr = (const float4*)(sD2 + j * 64);
        #pragma unroll
        for (int q = 0; q < 16; ++q) {
            const float4 wq = wr[q];
            d2[4*q+0] += dv * wq.x;
            d2[4*q+1] += dv * wq.y;
            d2[4*q+2] += dv * wq.z;
            d2[4*q+3] += dv * wq.w;
        }
    }

    // prob = sigmoid(leaky(d2) @ dec_w3 + db3)
    float logit = sDB3[0];
    {
        const float4* d3 = (const float4*)sD3;
        #pragma unroll
        for (int q = 0; q < 16; ++q) {
            const float4 wq = d3[q];
            logit += lrelu(d2[4*q+0]) * wq.x + lrelu(d2[4*q+1]) * wq.y
                   + lrelu(d2[4*q+2]) * wq.z + lrelu(d2[4*q+3]) * wq.w;
        }
    }
    const float p = 1.f / (1.f + expf(-logit));
    prob[row] = p;

    // deterministic per-block prob sum (fixed-order tree; no float atomics)
    __syncthreads();
    sRed[tid] = p;
    __syncthreads();
    for (int off = 128; off > 0; off >>= 1) {
        if (tid < off) sRed[tid] += sRed[tid + off];
        __syncthreads();
    }
    if (tid == 0) blocksum[blockIdx.x] = sRed[0];
}

// ---------------------------------------------------------------------------
// Kernel 3: water-fill v via bisection on f(v) = sum(min(p + v/2, 1)).
// One block, 1024 threads; each thread keeps 64 probs resident in VGPRs.
// ---------------------------------------------------------------------------
__global__ void __launch_bounds__(1024) waterfill_kernel(
    const float* __restrict__ prob, const float* __restrict__ blocksum,
    float* __restrict__ scal)
{
    __shared__ float red[1024];
    __shared__ float wsum[32];
    const int tid  = threadIdx.x;
    const int lane = tid & 31;
    const int wv   = tid >> 5;

    // deterministic total prob sum from 256 per-block sums
    red[tid] = (tid < 256) ? blocksum[tid] : 0.f;
    __syncthreads();
    for (int off = 512; off > 0; off >>= 1) {
        if (tid < off) red[tid] += red[tid + off];
        __syncthreads();
    }
    const float psum = red[0];
    __syncthreads();

    // load 64 probs per thread into registers
    const float4* p4 = (const float4*)prob;        // 16384 float4
    float4 r[16];
    #pragma unroll
    for (int j = 0; j < 16; ++j) r[j] = p4[tid + j * 1024];

    const float target = DELTA * (float)N_ROWS;
    float lo = -4.f, hi = 4.f;
    for (int it = 0; it < 32; ++it) {
        const float v  = 0.5f * (lo + hi);
        const float hv = 0.5f * v;
        float s = 0.f;
        #pragma unroll
        for (int j = 0; j < 16; ++j) {
            s += fminf(r[j].x + hv, 1.f) + fminf(r[j].y + hv, 1.f)
               + fminf(r[j].z + hv, 1.f) + fminf(r[j].w + hv, 1.f);
        }
        // wave reduce (no barrier), then cross-wave via LDS
        #pragma unroll
        for (int m = 16; m > 0; m >>= 1) s += __shfl_xor(s, m, 32);
        if (lane == 0) wsum[wv] = s;
        __syncthreads();
        float total = 0.f;
        #pragma unroll
        for (int w = 0; w < 32; ++w) total += wsum[w];
        __syncthreads();
        if (total >= target) hi = v; else lo = v;
    }
    if (tid == 0) {
        scal[1] = 0.5f * (lo + hi);                // v
        scal[2] = psum * (1.f / (float)N_ROWS);    // mean(prob)
    }
}

// ---------------------------------------------------------------------------
// Kernel 4: prob_out = mean >= DELTA ? p : min(p + v/2, 1)
// ---------------------------------------------------------------------------
__global__ void apply_kernel(const float* __restrict__ prob,
                             const float* __restrict__ scal,
                             float* __restrict__ out) {
    const int i = blockIdx.x * 256 + threadIdx.x;
    const float p = prob[i];
    const float v = scal[1];
    const float m = scal[2];
    out[i] = (m >= DELTA) ? p : fminf(p + 0.5f * v, 1.f);
}

// ---------------------------------------------------------------------------
extern "C" void kernel_launch(void* const* d_in, const int* in_sizes, int n_in,
                              void* d_out, int out_size, void* d_ws, size_t ws_size,
                              hipStream_t stream) {
    const float* x    = (const float*)d_in[0];
    const float* eps  = (const float*)d_in[1];
    const float* w1   = (const float*)d_in[2];
    const float* b1   = (const float*)d_in[3];
    const float* w2   = (const float*)d_in[4];
    const float* b2   = (const float*)d_in[5];
    const float* muw  = (const float*)d_in[6];
    const float* mub  = (const float*)d_in[7];
    const float* lvw  = (const float*)d_in[8];
    const float* lvb  = (const float*)d_in[9];
    const float* dw1  = (const float*)d_in[10];
    const float* db1  = (const float*)d_in[11];
    const float* dw2  = (const float*)d_in[12];
    const float* db2  = (const float*)d_in[13];
    const float* dw3  = (const float*)d_in[14];
    const float* db3  = (const float*)d_in[15];

    // workspace layout
    unsigned short* w1p   = (unsigned short*)d_ws;                    // 512 KB
    float* prob           = (float*)((char*)d_ws + 524288);           // 256 KB
    float* scal           = (float*)((char*)d_ws + 524288 + 262144);  // 32 B
    float* blocksum       = (float*)((char*)d_ws + 524288 + 262144 + 64); // 1 KB

    float* out = (float*)d_out;

    pack_w1_kernel<<<(F_DIM * H_DIM) / 256, 256, 0, stream>>>(w1, w1p, scal);
    vae_main_kernel<<<N_ROWS / 256, 256, 0, stream>>>(
        x, eps, w1p, b1, w2, b2, muw, mub, lvw, lvb,
        dw1, db1, dw2, db2, dw3, db3, out, prob, blocksum);
    waterfill_kernel<<<1, 1024, 0, stream>>>(prob, blocksum, scal);
    apply_kernel<<<N_ROWS / 256, 256, 0, stream>>>(prob, scal, out);
}
